// EdgeClassifier_50242527429292
// MI455X (gfx1250) — matrix operand order repaired and checked by reading, not compiled, not run on hardware
//
#include <hip/hip_runtime.h>
#include <hip/hip_bf16.h>

typedef __attribute__((ext_vector_type(16))) _Float16 v16h;
typedef __attribute__((ext_vector_type(8)))  float    v8f;

#define WAVES_PER_BLOCK 8
#define TPB (WAVES_PER_BLOCK * 32)
#define HSTRIDE 33   // 32 cols + 1 pad to spread LDS banks

__global__ __launch_bounds__(TPB) void edge_mlp_wmma_kernel(
    const float* __restrict__ nodeEmbed,   // [N,80]
    const int*   __restrict__ edge_index,  // [2,E]
    const float* __restrict__ W1,          // [161,32]
    const float* __restrict__ b1,          // [32]
    const float* __restrict__ W2,          // [32,32]
    const float* __restrict__ b2,          // [32]
    const float* __restrict__ W3,          // [32,2]
    const float* __restrict__ b3,          // [2]
    float*       __restrict__ out,         // [E,2]
    int E, int numTiles, int totalWaves)
{
    __shared__ float hst[WAVES_PER_BLOCK][16 * HSTRIDE];
    __shared__ float W3s[64];
    __shared__ float b3s[2];

    const int lane = threadIdx.x & 31;
    const int wid  = threadIdx.x >> 5;
    const int gwave = blockIdx.x * WAVES_PER_BLOCK + wid;
    const int col = lane & 15;   // B/C column selector, also A/M row selector
    const int hs  = lane >> 4;   // half-wave selector

    // One-time block staging of tiny layer-3 params (single barrier before loop).
    if (threadIdx.x < 64) W3s[threadIdx.x] = W3[threadIdx.x];
    if (threadIdx.x < 2)  b3s[threadIdx.x] = b3[threadIdx.x];
    __syncthreads();

    // ---- Per-wave resident weight fragments (f16, WMMA B layout) ----
    // B elem i <-> K = 16*hs + i ; lane holds column n = 16*nt + col.
    v16h b1f[6][2];
#pragma unroll
    for (int kc = 0; kc < 6; ++kc) {
#pragma unroll
        for (int nt = 0; nt < 2; ++nt) {
            v16h b;
#pragma unroll
            for (int i = 0; i < 16; ++i) {
                int c = kc * 32 + 16 * hs + i;           // row of W1 (K dim)
                float v = (c < 161) ? W1[c * 32 + nt * 16 + col] : 0.0f;
                b[i] = (_Float16)v;
            }
            b1f[kc][nt] = b;
        }
    }
    v16h b2f[2];
#pragma unroll
    for (int nt = 0; nt < 2; ++nt) {
        v16h b;
#pragma unroll
        for (int i = 0; i < 16; ++i) {
            int k = 16 * hs + i;
            b[i] = (_Float16)W2[k * 32 + nt * 16 + col];
        }
        b2f[nt] = b;
    }
    const float b1v0 = b1[col], b1v1 = b1[col + 16];
    const float b2v0 = b2[col], b2v1 = b2[col + 16];
    float* __restrict__ hw = hst[wid];

    // ---- Grid-stride over 16-edge tiles; one wave owns one tile ----
    for (int t = gwave; t < numTiles; t += totalWaves) {
        const int  row  = col;                       // this lane's edge row in tile
        const long eid  = (long)t * 16 + row;
        const long eidc = (eid < E) ? eid : (long)(E - 1);
        const int  sIdx = edge_index[eidc];
        const int  dIdx = edge_index[(size_t)E + eidc];
        const float* __restrict__ sp = nodeEmbed + (size_t)sIdx * 80;
        const float* __restrict__ dp = nodeEmbed + (size_t)dIdx * 80;

        // ---- Layer 1: [16x192(f16)] x [192x32] via 6 K-chunks x 2 N-tiles ----
        v8f acc0 = {0.f,0.f,0.f,0.f,0.f,0.f,0.f,0.f};
        v8f acc1 = {0.f,0.f,0.f,0.f,0.f,0.f,0.f,0.f};
#pragma unroll
        for (int kc = 0; kc < 6; ++kc) {
            v16h a;
#pragma unroll
            for (int i = 0; i < 16; ++i) {
                // A elem i <-> K = 16*hs+i
                int krel = 16 * hs + i;
                int c = kc * 32 + krel;              // column of edge embedding
                float v = 0.0f;
                if (c < 80)                 v = sp[c];
                else if (c == 80)           v = 1.0f;
                else if (c < 161)           v = dp[c - 81];
                a[i] = (_Float16)v;
            }
            acc0 = __builtin_amdgcn_wmma_f32_16x16x32_f16(
                       false, a, false, b1f[kc][0], (short)0, acc0, false, false);
            acc1 = __builtin_amdgcn_wmma_f32_16x16x32_f16(
                       false, a, false, b1f[kc][1], (short)0, acc1, false, false);
        }

        // h1 = relu(acc + b1) -> LDS row-major [16][32] (per-wave buffer)
        // D layout: VGPR r holds M = r + 8*hs, N = col (+16 for acc1)
#pragma unroll
        for (int r = 0; r < 8; ++r) {
            int m = r + 8 * hs;
            hw[m * HSTRIDE + col]      = fmaxf(acc0[r] + b1v0, 0.0f);
            hw[m * HSTRIDE + col + 16] = fmaxf(acc1[r] + b1v1, 0.0f);
        }
        __asm__ volatile("s_wait_dscnt 0" ::: "memory");  // cross-lane LDS RAW

        // ---- Layer 2: transpose-read h1 as A fragment (K=32, one chunk) ----
        v16h a2;
#pragma unroll
        for (int i = 0; i < 16; ++i) {
            int krel = 16 * hs + i;
            a2[i] = (_Float16)hw[row * HSTRIDE + krel];
        }
        v8f c20 = {0.f,0.f,0.f,0.f,0.f,0.f,0.f,0.f};
        v8f c21 = {0.f,0.f,0.f,0.f,0.f,0.f,0.f,0.f};
        c20 = __builtin_amdgcn_wmma_f32_16x16x32_f16(
                  false, a2, false, b2f[0], (short)0, c20, false, false);
        c21 = __builtin_amdgcn_wmma_f32_16x16x32_f16(
                  false, a2, false, b2f[1], (short)0, c21, false, false);

        // h2 = relu(c2 + b2) -> LDS (WAR vs a2 loads is safe: LDS is in-order)
#pragma unroll
        for (int r = 0; r < 8; ++r) {
            int m = r + 8 * hs;
            hw[m * HSTRIDE + col]      = fmaxf(c20[r] + b2v0, 0.0f);
            hw[m * HSTRIDE + col + 16] = fmaxf(c21[r] + b2v1, 0.0f);
        }
        __asm__ volatile("s_wait_dscnt 0" ::: "memory");  // cross-lane LDS RAW

        // ---- Layer 3 + log_softmax (tiny: VALU, lanes 0-15 write) ----
        float l0 = b3s[0], l1 = b3s[1];
#pragma unroll
        for (int k = 0; k < 32; ++k) {
            float h = hw[row * HSTRIDE + k];
            l0 = fmaf(h, W3s[2 * k + 0], l0);
            l1 = fmaf(h, W3s[2 * k + 1], l1);
        }
        float mx  = fmaxf(l0, l1);
        float lse = mx + __logf(__expf(l0 - mx) + __expf(l1 - mx));
        if (hs == 0 && eid < (long)E) {
            float2 o = make_float2(l0 - lse, l1 - lse);
            *(float2*)(out + (size_t)eid * 2) = o;
        }
    }
}

extern "C" void kernel_launch(void* const* d_in, const int* in_sizes, int n_in,
                              void* d_out, int out_size, void* d_ws, size_t ws_size,
                              hipStream_t stream) {
    const float* nodeEmbed  = (const float*)d_in[0];
    const int*   edge_index = (const int*)  d_in[1];
    const float* W1 = (const float*)d_in[2];
    const float* b1 = (const float*)d_in[3];
    const float* W2 = (const float*)d_in[4];
    const float* b2 = (const float*)d_in[5];
    const float* W3 = (const float*)d_in[6];
    const float* b3 = (const float*)d_in[7];
    float* out = (float*)d_out;

    const int E        = in_sizes[1] / 2;          // edge_index is [2,E]
    const int numTiles = (E + 15) / 16;
    int blocks = (numTiles + WAVES_PER_BLOCK - 1) / WAVES_PER_BLOCK;
    if (blocks > 4096) blocks = 4096;              // grid-stride beyond this
    const int totalWaves = blocks * WAVES_PER_BLOCK;

    edge_mlp_wmma_kernel<<<blocks, TPB, 0, stream>>>(
        nodeEmbed, edge_index, W1, b1, W2, b2, W3, b3, out,
        E, numTiles, totalWaves);
}
